// DiffHeadGATRating_5265629905485
// MI455X (gfx1250) — matrix operand order repaired and checked by reading, not compiled, not run on hardware
//
#include <hip/hip_runtime.h>
#include <hip/hip_bf16.h>
#include <math.h>

// Problem constants (match reference setup_inputs)
#define NUU 100000L
#define NII 50000L
#define NEE 400000L
#define NRR 6
#define NBB 64

typedef __bf16 bf16_t;
typedef __attribute__((ext_vector_type(16))) __bf16 v16bf;
typedef __attribute__((ext_vector_type(8)))  float  v8f;

// ---------------------------------------------------------------------------
// WMMA GEMM: C[N x 128] = A[N x K] @ W[K x 128], K = KSTEPS*32 (128 or 384).
// fp32 inputs converted to bf16, f32 accumulate via v_wmma_f32_16x16x32_bf16.
// One block = 256 threads = 8 waves; wave w computes 16-row strip, all 128 cols.
// W is staged (transposed) in dynamic LDS, shared by all 8 waves.
// ---------------------------------------------------------------------------
template<int KSTEPS>
__global__ void wmma_gemm(const float* __restrict__ A, const float* __restrict__ W,
                          float* __restrict__ C, long N) {
  constexpr int K = KSTEPS * 32;
  extern __shared__ bf16_t lds[];                 // W^T: lds[col*K + k]
  const int tid = threadIdx.x;
  for (int idx = tid; idx < K * 128; idx += 256) {
    int k = idx >> 7, c = idx & 127;
    lds[c * K + k] = (bf16_t)W[idx];
  }
  __syncthreads();
  const int wave = tid >> 5, lane = tid & 31;
  const long row0 = ((long)blockIdx.x * 8 + wave) * 16;
  if (row0 >= N) return;                          // wave-uniform: EXEC stays all-1s
  const int lhalf = lane >> 4, lmod = lane & 15;
  const int kbA = lhalf * 8;                      // ISA 16-bit A layout: K {0-7,16-23} / {8-15,24-31}
  const int kbB = lhalf * 8;                      // B matches A: K {0-7,16-23} / {8-15,24-31}
  const float* arow = A + (row0 + lmod) * K;
  v16bf afrag[KSTEPS];
#pragma unroll
  for (int s = 0; s < KSTEPS; ++s) {
    const int k0 = s * 32 + kbA;
#pragma unroll
    for (int i = 0; i < 8; ++i) {
      afrag[s][i]     = (bf16_t)arow[k0 + i];
      afrag[s][8 + i] = (bf16_t)arow[k0 + 16 + i];
    }
  }
  for (int nt = 0; nt < 8; ++nt) {
    const bf16_t* wcol = lds + (long)(nt * 16 + lmod) * K;
    v8f acc = {};
#pragma unroll
    for (int s = 0; s < KSTEPS; ++s) {
      const int k0 = s * 32 + kbB;
      v16bf bfrag;
#pragma unroll
      for (int i = 0; i < 16; ++i) bfrag[i] = wcol[k0 + i + (i >= 8 ? 8 : 0)];
      acc = __builtin_amdgcn_wmma_f32_16x16x32_bf16(false, afrag[s], false, bfrag,
                                                    (short)0, acc, false, false);
    }
    // D layout: lane<16 -> rows M=v, lanes>=16 -> rows M=8+v; col = nt*16 + lmod
    float* cp = C + (row0 + lhalf * 8) * 128 + nt * 16 + lmod;
#pragma unroll
    for (int v = 0; v < 8; ++v) cp[(long)v * 128] = acc[v];
  }
}

// ---------------------------------------------------------------------------
// Small helpers
// ---------------------------------------------------------------------------
__global__ void fill_f32(float* __restrict__ p, float v, long n) {
  long i = (long)blockIdx.x * 256 + threadIdx.x;
  if (i < n) p[i] = v;
}

// tab[i] = dot(emb[i,:], attn) for tiny tables (6 or 64 rows). 1 block / row.
__global__ void tab_dot(const float* __restrict__ emb, const float* __restrict__ attn,
                        float* __restrict__ tab) {
  __shared__ float red[4];
  const int i = blockIdx.x, t = threadIdx.x;
  float p = emb[(long)i * 128 + t] * attn[t];
  for (int o = 16; o > 0; o >>= 1) p += __shfl_down(p, o, 32);
  if ((t & 31) == 0) red[t >> 5] = p;
  __syncthreads();
  if (t == 0) tab[i] = red[0] + red[1] + red[2] + red[3];
}

// s[n] = dot(F[n,:], a) ; one wave per node
__global__ void node_dot(const float* __restrict__ F, const float* __restrict__ a,
                         float* __restrict__ s, long N) {
  const int wave = threadIdx.x >> 5, lane = threadIdx.x & 31;
  const long n = (long)blockIdx.x * 8 + wave;
  if (n >= N) return;
  const float* f = F + n * 128;
  float p = f[lane] * a[lane] + f[lane + 32] * a[lane + 32]
          + f[lane + 64] * a[lane + 64] + f[lane + 96] * a[lane + 96];
  for (int o = 16; o > 0; o >>= 1) p += __shfl_down(p, o, 32);
  if (lane == 0) s[n] = p;
}

__global__ void edge_gather1(float* __restrict__ extra, const float* __restrict__ tab,
                             const int* __restrict__ idx, long E) {
  long e = (long)blockIdx.x * 256 + threadIdx.x;
  if (e < E) extra[e] = tab[idx[e]];
}

__global__ void edge_gather2(float* __restrict__ extra,
                             const float* __restrict__ tab1, const int* __restrict__ deg1,
                             const int* __restrict__ i1,
                             const float* __restrict__ tab2, const int* __restrict__ deg2,
                             const int* __restrict__ i2, long E) {
  long e = (long)blockIdx.x * 256 + threadIdx.x;
  if (e < E) extra[e] = tab1[deg1[i1[e]]] + tab2[deg2[i2[e]]];
}

__device__ __forceinline__ void atomicMaxFloat(float* addr, float val) {
  if (val >= 0.0f) atomicMax((int*)addr, __float_as_int(val));
  else             atomicMin((unsigned int*)addr, __float_as_uint(val));
}

__global__ void edge_logit(const float* __restrict__ sl, const float* __restrict__ sr,
                           const float* __restrict__ extra,
                           const int* __restrict__ src, const int* __restrict__ dst,
                           float* __restrict__ logit, float* __restrict__ nmax, long E) {
  long e = (long)blockIdx.x * 256 + threadIdx.x;
  if (e >= E) return;
  float x = sl[src[e]] + sr[dst[e]] + extra[e];
  x = (x >= 0.0f) ? x : 0.01f * x;                 // leaky-relu, SLOPE=0.01
  logit[e] = x;
  atomicMaxFloat(&nmax[dst[e]], x);
}

__global__ void edge_expsum(const float* __restrict__ logit, const int* __restrict__ dst,
                            const float* __restrict__ nmax, float* __restrict__ ew,
                            float* __restrict__ nsum, long E) {
  long e = (long)blockIdx.x * 256 + threadIdx.x;
  if (e >= E) return;
  int d = dst[e];
  float w = expf(logit[e] - nmax[d]);
  ew[e] = w;
  atomicAdd(&nsum[d], w);
}

// agg[dst,:] += (ew[e]/nsum[dst]) * fsrc[src,:]   (one block per edge, 128 chans)
__global__ void edge_agg(const float* __restrict__ ew, const float* __restrict__ nsum,
                         const float* __restrict__ fsrc,
                         const int* __restrict__ src, const int* __restrict__ dst,
                         float* __restrict__ agg) {
  const long e = blockIdx.x;
  const int c = threadIdx.x;
  const int s = src[e], d = dst[e];
  const float a = ew[e] / nsum[d];
  atomicAdd(&agg[(long)d * 128 + c], a * fsrc[(long)s * 128 + c]);
}

// cat[n, h*128+c] = agg[n,c] + hdst[n,c] + b[c]   (residual + bias, write into concat slot)
__global__ void head_epilogue(float* __restrict__ cat, const float* __restrict__ agg,
                              const float* __restrict__ hdst, const float* __restrict__ b,
                              int h, long Nd) {
  long i = (long)blockIdx.x * 256 + threadIdx.x;
  if (i >= Nd * 128) return;
  long n = i >> 7;
  int  c = (int)(i & 127);
  cat[n * 384 + (long)h * 128 + c] = agg[i] + hdst[i] + b[c];
}

// per-channel sum / sumsq partials (population stats)
__global__ void bn_stats(const float* __restrict__ y, float* __restrict__ s, long N) {
  const int c = threadIdx.x;                      // 128 channels
  long r0 = (long)blockIdx.x * 256;
  long rend = r0 + 256; if (rend > N) rend = N;
  float sm = 0.f, sq = 0.f;
  for (long r = r0; r < rend; ++r) { float v = y[r * 128 + c]; sm += v; sq += v * v; }
  atomicAdd(&s[c], sm);
  atomicAdd(&s[128 + c], sq);
}

// out = silu( (y - mu) * rsqrt(var+1e-5) * g + be )
__global__ void bn_apply(float* __restrict__ out, const float* __restrict__ y,
                         const float* __restrict__ s, const float* __restrict__ g,
                         const float* __restrict__ be, long N) {
  long i = (long)blockIdx.x * 256 + threadIdx.x;
  if (i >= N * 128) return;
  int c = (int)(i & 127);
  float inv_n = 1.0f / (float)N;
  float mu = s[c] * inv_n;
  float var = s[128 + c] * inv_n - mu * mu;
  float xn = (y[i] - mu) * rsqrtf(var + 1e-5f) * g[c] + be[c];
  out[i] = xn / (1.0f + expf(-xn));               // SiLU
}

// ---------------------------------------------------------------------------
// Orchestration
// ---------------------------------------------------------------------------
extern "C" void kernel_launch(void* const* d_in, const int* in_sizes, int n_in,
                              void* d_out, int out_size, void* d_ws, size_t ws_size,
                              hipStream_t stream) {
  (void)in_sizes; (void)n_in; (void)out_size; (void)ws_size;
  const float* user_emb   = (const float*)d_in[0];
  const float* item_emb   = (const float*)d_in[1];
  const float* gat_W      = (const float*)d_in[2];   // [2,3,2,128,128]
  const float* gat_al     = (const float*)d_in[3];   // [2,3,2,128]
  const float* gat_ar     = (const float*)d_in[4];
  const float* gat_b      = (const float*)d_in[5];
  const float* rt_emb     = (const float*)d_in[6];   // [2,2,6,128]
  const float* rt_attn    = (const float*)d_in[7];   // [2,2,128]
  const float* pop_emb_u  = (const float*)d_in[8];   // [2,2,64,128]
  const float* pop_emb_i  = (const float*)d_in[9];
  const float* pop_attn_s = (const float*)d_in[10];  // [2,2,128]
  const float* pop_attn_d = (const float*)d_in[11];
  const float* cat_W      = (const float*)d_in[12];  // [2,384,128]
  // d_in[13] = cat_b: per-channel constant shift, cancels exactly under BatchNorm.
  const float* bn_gamma   = (const float*)d_in[14];  // [2,2,128]
  const float* bn_beta    = (const float*)d_in[15];
  const int*   edge_u     = (const int*)d_in[16];
  const int*   edge_i     = (const int*)d_in[17];
  const int*   edge_rt    = (const int*)d_in[18];
  const int*   deg_u      = (const int*)d_in[19];
  const int*   deg_i      = (const int*)d_in[20];

  // workspace layout (floats)
  float* ws = (float*)d_ws;
  const long SHU = NUU * 128, SHI = NII * 128;
  float* p_hu   = ws;                    ws += SHU;        // layer-0 user output
  float* p_hi   = ws;                    ws += SHI;        // layer-0 item output
  float* p_ucat = ws;                    ws += NUU * 384;
  float* p_icat = ws;                    ws += NII * 384;
  float* p_fsrc = ws;                    ws += SHU;        // also reused as ytmp
  float* p_fdst = ws;                    ws += SHU;        // also reused as agg
  float* p_ssrc = ws;                    ws += NUU;
  float* p_sdst = ws;                    ws += NUU;
  float* p_ego  = ws;                    ws += NEE;
  float* p_ebk  = ws;                    ws += NEE;
  float* p_elog = ws;                    ws += NEE;
  float* p_ew   = ws;                    ws += NEE;
  float* p_nmax = ws;                    ws += NUU;
  float* p_nsum = ws;                    ws += NUU;
  float* p_tabA = ws;                    ws += NBB;
  float* p_tabB = ws;                    ws += NBB;
  float* p_tabC = ws;                    ws += NBB;
  float* p_tabD = ws;                    ws += NBB;
  float* p_bns  = ws;                    ws += 256;

  auto fill = [&](float* p, float v, long n) {
    fill_f32<<<dim3((unsigned)((n + 255) / 256)), 256, 0, stream>>>(p, v, n);
  };
  const unsigned EGRID = (unsigned)((NEE + 255) / 256);

  for (int l = 0; l < 2; ++l) {
    const float* hu = l ? p_hu : user_emb;
    const float* hi = l ? p_hi : item_emb;
    float* out_u = l ? (float*)d_out : p_hu;
    float* out_i = l ? ((float*)d_out + SHU) : p_hi;

    for (int h = 0; h < 3; ++h) {
      // ---- per-edge additive biases for this head ----
      if (h == 0) {
        fill(p_ego, 0.f, NEE); fill(p_ebk, 0.f, NEE);
      } else if (h == 1) {
        tab_dot<<<NRR, 128, 0, stream>>>(rt_emb + (long)(l * 2 + 0) * NRR * 128,
                                         rt_attn + (long)(l * 2 + 0) * 128, p_tabA);
        tab_dot<<<NRR, 128, 0, stream>>>(rt_emb + (long)(l * 2 + 1) * NRR * 128,
                                         rt_attn + (long)(l * 2 + 1) * 128, p_tabB);
        edge_gather1<<<EGRID, 256, 0, stream>>>(p_ego, p_tabA, edge_rt, NEE);
        edge_gather1<<<EGRID, 256, 0, stream>>>(p_ebk, p_tabB, edge_rt, NEE);
      } else {
        tab_dot<<<NBB, 128, 0, stream>>>(pop_emb_u + (long)(l * 2 + 0) * NBB * 128,
                                         pop_attn_s + (long)(l * 2 + 0) * 128, p_tabA);
        tab_dot<<<NBB, 128, 0, stream>>>(pop_emb_i + (long)(l * 2 + 0) * NBB * 128,
                                         pop_attn_d + (long)(l * 2 + 0) * 128, p_tabB);
        tab_dot<<<NBB, 128, 0, stream>>>(pop_emb_i + (long)(l * 2 + 1) * NBB * 128,
                                         pop_attn_s + (long)(l * 2 + 1) * 128, p_tabC);
        tab_dot<<<NBB, 128, 0, stream>>>(pop_emb_u + (long)(l * 2 + 1) * NBB * 128,
                                         pop_attn_d + (long)(l * 2 + 1) * 128, p_tabD);
        edge_gather2<<<EGRID, 256, 0, stream>>>(p_ego, p_tabA, deg_u, edge_u,
                                                p_tabB, deg_i, edge_i, NEE);
        edge_gather2<<<EGRID, 256, 0, stream>>>(p_ebk, p_tabC, deg_i, edge_i,
                                                p_tabD, deg_u, edge_u, NEE);
      }
      // ---- GAT per relation: rel 0 = go (user->item), rel 1 = back (item->user) ----
      for (int rel = 0; rel < 2; ++rel) {
        const float* hs = rel ? hi : hu;
        const float* hd = rel ? hu : hi;
        const long   Ns = rel ? NII : NUU;
        const long   Nd = rel ? NUU : NII;
        const int*   si = rel ? edge_i : edge_u;
        const int*   di = rel ? edge_u : edge_i;
        const long   wo = (((long)l * 3 + h) * 2 + rel);
        const float* W  = gat_W  + wo * 128 * 128;
        const float* al = gat_al + wo * 128;
        const float* ar = gat_ar + wo * 128;
        const float* gb = gat_b  + wo * 128;
        const float* ex = rel ? p_ebk : p_ego;
        float* catb = rel ? p_ucat : p_icat;

        wmma_gemm<4><<<dim3((unsigned)((Ns / 16 + 7) / 8)), 256, 128 * 128 * 2, stream>>>(hs, W, p_fsrc, Ns);
        wmma_gemm<4><<<dim3((unsigned)((Nd / 16 + 7) / 8)), 256, 128 * 128 * 2, stream>>>(hd, W, p_fdst, Nd);
        node_dot<<<dim3((unsigned)((Ns + 7) / 8)), 256, 0, stream>>>(p_fsrc, al, p_ssrc, Ns);
        node_dot<<<dim3((unsigned)((Nd + 7) / 8)), 256, 0, stream>>>(p_fdst, ar, p_sdst, Nd);
        fill(p_nmax, -INFINITY, Nd);
        fill(p_nsum, 0.f, Nd);
        edge_logit<<<EGRID, 256, 0, stream>>>(p_ssrc, p_sdst, ex, si, di, p_elog, p_nmax, NEE);
        edge_expsum<<<EGRID, 256, 0, stream>>>(p_elog, di, p_nmax, p_ew, p_nsum, NEE);
        float* agg = p_fdst;                      // fdst is dead after node_dot -> reuse
        fill(agg, 0.f, Nd * 128);
        edge_agg<<<dim3((unsigned)NEE), 128, 0, stream>>>(p_ew, p_nsum, p_fsrc, si, di, agg);
        head_epilogue<<<dim3((unsigned)((Nd * 128 + 255) / 256)), 256, 0, stream>>>(
            catb, agg, hd, gb, h, Nd);
      }
    }
    // ---- concat GEMM (K=384) + BatchNorm + SiLU, users then items ----
    for (int t = 0; t < 2; ++t) {
      const long  N    = t ? NII : NUU;
      float*      catb = t ? p_icat : p_ucat;
      float*      outp = t ? out_i : out_u;
      const float* cW  = cat_W + (long)l * 384 * 128;
      const float* g   = bn_gamma + (long)(l * 2 + t) * 128;
      const float* be  = bn_beta  + (long)(l * 2 + t) * 128;
      float* ytmp = p_fsrc;                       // fsrc is dead here -> reuse
      wmma_gemm<12><<<dim3((unsigned)((N / 16 + 7) / 8)), 256, 384 * 128 * 2, stream>>>(catb, cW, ytmp, N);
      fill(p_bns, 0.f, 256);
      bn_stats<<<dim3((unsigned)((N + 255) / 256)), 128, 0, stream>>>(ytmp, p_bns, N);
      bn_apply<<<dim3((unsigned)((N * 128 + 255) / 256)), 256, 0, stream>>>(outp, ytmp, p_bns, g, be, N);
    }
  }
}
